// FlashScpFormerMHA_3178275799748
// MI455X (gfx1250) — matrix operand order repaired and checked by reading, not compiled, not run on hardware
//
#include <hip/hip_runtime.h>
#include <math.h>

// ---------------------------------------------------------------------------
// Problem constants (from the reference)
// ---------------------------------------------------------------------------
#define B_    4
#define CTX_  2048
#define TGT_  512
#define E_    1024
#define H_    16
#define D_    64
#define S_TOT (CTX_ + TGT_)      // 2560
#define NEGMASK (-1.0e9f)
#define NEGINF  (-1.0e30f)
#define SCALE   (0.125f)         // 1/sqrt(64)

typedef __bf16 bf16_t;
typedef __attribute__((ext_vector_type(16))) __bf16 bf16x16;
typedef __attribute__((ext_vector_type(8)))  __bf16 bf16x8;
typedef __attribute__((ext_vector_type(4)))  __bf16 bf16x4;
typedef __attribute__((ext_vector_type(8)))  float  floatx8;

union FragU { bf16x16 v; bf16x8 h[2]; };

__device__ __forceinline__ floatx8 wmma_bf16(bf16x16 a, bf16x16 b, floatx8 c) {
  // D(f32 16x16) = A(bf16 16x32) * B(bf16 32x16) + C
  return __builtin_amdgcn_wmma_f32_16x16x32_bf16(false, a, false, b,
                                                 (short)0, c, false, false);
}

// A-fragment from a row-major (M x >=32) bf16 LDS tile.
// ISA layout: lane m = l&15; lanes 0-15 hold K 0-7 & 16-23, lanes 16-31 hold
// K 8-15 & 24-31  ->  two contiguous ds_load_b128 per lane.
__device__ __forceinline__ bf16x16 ld_frag_a(const bf16_t* tile, int stride) {
  const int lane = threadIdx.x & 31;
  const bf16_t* p = tile + (lane & 15) * stride + ((lane >> 4) << 3);
  FragU u;
  u.h[0] = *(const bf16x8*)(p);
  u.h[1] = *(const bf16x8*)(p + 16);
  return u.v;
}

// B-fragment: tile stored column(n)-major with the K dim contiguous
// (row n of the LDS tile = column n of B, length >= 32 bf16).
// lanes 0-15 hold K 0-7 & 16-23, lanes 16-31 hold K 8-15 & 24-31.
__device__ __forceinline__ bf16x16 ld_frag_b(const bf16_t* tile, int stride) {
  const int lane = threadIdx.x & 31;
  const bf16_t* p = tile + (lane & 15) * stride + ((lane >> 4) << 3);
  FragU u;
  u.h[0] = *(const bf16x8*)(p);
  u.h[1] = *(const bf16x8*)(p + 16);
  return u.v;
}

__device__ __forceinline__ void st4(bf16_t* p, float4 v) {
  bf16x4 t;
  t[0] = (__bf16)v.x; t[1] = (__bf16)v.y; t[2] = (__bf16)v.z; t[3] = (__bf16)v.w;
  *(bf16x4*)p = t;
}

__device__ __forceinline__ bf16x8 cvt8(const float* p) {
  float4 a = *(const float4*)p;
  float4 b = *(const float4*)(p + 4);
  bf16x8 r;
  r[0] = (__bf16)a.x; r[1] = (__bf16)a.y; r[2] = (__bf16)a.z; r[3] = (__bf16)a.w;
  r[4] = (__bf16)b.x; r[5] = (__bf16)b.y; r[6] = (__bf16)b.z; r[7] = (__bf16)b.w;
  return r;
}

// Reductions across a 16-lane half of the wave (xor masks 1..8 stay in-half).
__device__ __forceinline__ float half_max(float v) {
  #pragma unroll
  for (int m = 1; m <= 8; m <<= 1) v = fmaxf(v, __shfl_xor(v, m, 32));
  return v;
}
__device__ __forceinline__ float half_sum(float v) {
  #pragma unroll
  for (int m = 1; m <= 8; m <<= 1) v += __shfl_xor(v, m, 32);
  return v;
}

// ---------------------------------------------------------------------------
// Generic GEMM:  C[m, n] = sum_k Arow(m)[k] * W[n, k] + bias[n]
// A rows are gathered through a two-segment (per-batch concat) scheme:
//   per = seg1 + seg2 ; b = m / per ; s = m % per
//   s <  seg1 : row = A1 + (b*seg1 + s)        * lda1
//   s >= seg1 : row = A2 + (b*seg2 + s - seg1) * lda2
// (plain GEMM: seg1 = M_total, seg2 = 0)
//
// Block tile 128(M) x 256(N) x 32(K); 256 threads = 8 waves in a 2(M) x 4(N)
// grid; each wave owns a 64x64 tile -> 16 WMMAs per k-step against only
// 8 fragment loads.  LDS is double-buffered: the next k-step's
// global_load_b128s are issued before the WMMAs and stored to the alternate
// buffer, with a single s_barrier per iteration.
// ---------------------------------------------------------------------------
__global__ __launch_bounds__(256) void gemm_bias_kernel(
    const float* __restrict__ A1, long lda1, int seg1,
    const float* __restrict__ A2, long lda2, int seg2,
    const float* __restrict__ W,  const float* __restrict__ bias,
    float* __restrict__ C, long ldc, int K)
{
  __shared__ bf16_t sA[2][128 * 32];   // 8 KB per buffer
  __shared__ bf16_t sB[2][256 * 32];   // 16 KB per buffer

  const int tid  = threadIdx.x;
  const int lane = tid & 31;
  const int wave = tid >> 5;
  const int wm   = wave & 1;          // 2 waves along M (64 rows each)
  const int wn   = wave >> 1;         // 4 waves along N (64 cols each)
  const int m0   = blockIdx.y * 128;
  const int n0   = blockIdx.x * 256;
  const int per  = seg1 + seg2;

  // Staging plan: A = 1024 float4 (4/thread), B = 2048 float4 (8/thread).
  const float* arow[4];
  int aoff[4];
  #pragma unroll
  for (int i = 0; i < 4; ++i) {
    int vec = tid + i * 256;          // 0..1023
    int r   = vec >> 3;               // 0..127
    int c   = (vec & 7) << 2;         // 0,4,..,28
    aoff[i] = r * 32 + c;
    int m  = m0 + r;
    int bb = m / per;
    int s  = m - bb * per;
    arow[i] = (s < seg1) ? (A1 + (long)(bb * seg1 + s) * lda1 + c)
                         : (A2 + (long)(bb * seg2 + (s - seg1)) * lda2 + c);
  }
  const float* wrow[8];
  int woff[8];
  #pragma unroll
  for (int i = 0; i < 8; ++i) {
    int vec = tid + i * 256;          // 0..2047
    int r   = vec >> 3;               // 0..255
    int c   = (vec & 7) << 2;
    woff[i] = r * 32 + c;
    wrow[i] = W + (long)(n0 + r) * K + c;
  }

  floatx8 acc[4][4];
  #pragma unroll
  for (int mi = 0; mi < 4; ++mi)
    #pragma unroll
    for (int ni = 0; ni < 4; ++ni)
      acc[mi][ni] = (floatx8){0.f,0.f,0.f,0.f,0.f,0.f,0.f,0.f};

  // ---- pipeline prologue: tile 0 -> regs -> LDS buf 0 ----
  float4 av[4], wv[8];
  #pragma unroll
  for (int i = 0; i < 4; ++i) av[i] = *(const float4*)(arow[i]);
  #pragma unroll
  for (int i = 0; i < 8; ++i) wv[i] = *(const float4*)(wrow[i]);
  #pragma unroll
  for (int i = 0; i < 4; ++i) st4(&sA[0][aoff[i]], av[i]);
  #pragma unroll
  for (int i = 0; i < 8; ++i) st4(&sB[0][woff[i]], wv[i]);
  __syncthreads();

  const int nIter = K >> 5;
  for (int it = 0; it < nIter; ++it) {
    const int cur = it & 1;
    const bool more = (it + 1) < nIter;
    if (more) {                        // issue next tile's loads up front
      const int k = (it + 1) << 5;
      #pragma unroll
      for (int i = 0; i < 4; ++i) av[i] = *(const float4*)(arow[i] + k);
      #pragma unroll
      for (int i = 0; i < 8; ++i) wv[i] = *(const float4*)(wrow[i] + k);
      __builtin_prefetch(arow[0] + k + 32, 0, 1);
      __builtin_prefetch(wrow[0] + k + 32, 0, 1);
    }

    // 16 WMMAs from the current buffer (loads overlap these).
    bf16x16 af[4], bfr[4];
    #pragma unroll
    for (int mi = 0; mi < 4; ++mi)
      af[mi] = ld_frag_a(sA[cur] + (wm * 64 + mi * 16) * 32, 32);
    #pragma unroll
    for (int ni = 0; ni < 4; ++ni)
      bfr[ni] = ld_frag_b(sB[cur] + (wn * 64 + ni * 16) * 32, 32);
    #pragma unroll
    for (int mi = 0; mi < 4; ++mi)
      #pragma unroll
      for (int ni = 0; ni < 4; ++ni)
        acc[mi][ni] = wmma_bf16(af[mi], bfr[ni], acc[mi][ni]);

    if (more) {                        // store to the *other* buffer
      #pragma unroll
      for (int i = 0; i < 4; ++i) st4(&sA[cur ^ 1][aoff[i]], av[i]);
      #pragma unroll
      for (int i = 0; i < 8; ++i) st4(&sB[cur ^ 1][woff[i]], wv[i]);
    }
    __syncthreads();
  }

  // Epilogue: C layout -> register r holds row (r + 8*laneHalf), lane = col.
  const int colL = lane & 15;
  const int hi   = lane >> 4;
  #pragma unroll
  for (int mi = 0; mi < 4; ++mi) {
    #pragma unroll
    for (int ni = 0; ni < 4; ++ni) {
      int row0 = m0 + wm * 64 + mi * 16 + hi * 8;
      int col  = n0 + wn * 64 + ni * 16 + colL;
      float bv = bias[col];
      #pragma unroll
      for (int r = 0; r < 8; ++r)
        C[(long)(row0 + r) * ldc + col] = acc[mi][ni][r] + bv;
    }
  }
}

// ---------------------------------------------------------------------------
// Flash self-attention over ctx.  One wave = one 16-row Q tile; all 8 waves
// of the block share the streamed 32-key K/V LDS chunks.
// qkv: [B*CTX, 3E] (q|k|v column slices), out: [B*CTX, E] (cols = h*D + d)
// ---------------------------------------------------------------------------
__global__ __launch_bounds__(256) void flash_self_kernel(
    const float* __restrict__ qkv,
    const unsigned char* __restrict__ cmask,
    float* __restrict__ outp)
{
  __shared__ bf16_t sK[32 * 64];       // [key][d]
  __shared__ bf16_t sV[64 * 32];       // transposed: [d][key]
  __shared__ bf16_t sP[8][16 * 32];    // per-wave P relayout scratch

  const int tid  = threadIdx.x;
  const int lane = tid & 31;
  const int wave = tid >> 5;
  const int h = blockIdx.y, b = blockIdx.z;
  const int q0 = blockIdx.x * 128 + wave * 16;
  const long ld = 3 * E_;
  const float* qbase = qkv + (long)b * CTX_ * ld + h * D_;
  const float* kbase = qbase + E_;
  const float* vbase = qbase + 2 * E_;
  const unsigned char* mk = cmask + b * CTX_;

  // Q fragments: 16x64 -> two 16x32 A-frags, loaded straight from global.
  bf16x16 qa[2];
  {
    const int m = lane & 15, kh = lane >> 4;
    const float* qr = qbase + (long)(q0 + m) * ld;
    #pragma unroll
    for (int c = 0; c < 2; ++c) {
      FragU u;
      u.h[0] = cvt8(qr + c * 32 + kh * 8);
      u.h[1] = cvt8(qr + c * 32 + kh * 8 + 16);
      qa[c] = u.v;
    }
  }

  floatx8 o[4];
  #pragma unroll
  for (int dt = 0; dt < 4; ++dt)
    o[dt] = (floatx8){0.f,0.f,0.f,0.f,0.f,0.f,0.f,0.f};
  float mrun[8], lrun[8];
  #pragma unroll
  for (int r = 0; r < 8; ++r) { mrun[r] = NEGINF; lrun[r] = 0.f; }

  for (int kc = 0; kc < CTX_; kc += 32) {
    __syncthreads();
    #pragma unroll
    for (int i = 0; i < 2; ++i) {      // stage 32x64 K and V (transposed)
      int vec = tid + i * 256;         // 0..511
      int r   = vec >> 4;              // key 0..31
      int c   = (vec & 15) << 2;       // d 0..60
      float4 kv = *(const float4*)(kbase + (long)(kc + r) * ld + c);
      st4(&sK[r * 64 + c], kv);
      float4 vv = *(const float4*)(vbase + (long)(kc + r) * ld + c);
      sV[(c + 0) * 32 + r] = (__bf16)vv.x;
      sV[(c + 1) * 32 + r] = (__bf16)vv.y;
      sV[(c + 2) * 32 + r] = (__bf16)vv.z;
      sV[(c + 3) * 32 + r] = (__bf16)vv.w;
    }
    __syncthreads();

    floatx8 s[2];
    #pragma unroll
    for (int kt = 0; kt < 2; ++kt) {   // scores: Q(16x64) . K^T -> 16x16
      floatx8 a = (floatx8){0.f,0.f,0.f,0.f,0.f,0.f,0.f,0.f};
      a = wmma_bf16(qa[0], ld_frag_b(sK + kt * 16 * 64,      64), a);
      a = wmma_bf16(qa[1], ld_frag_b(sK + kt * 16 * 64 + 32, 64), a);
      s[kt] = a;
    }
    const bool kval0 = mk[kc +      (lane & 15)] != 0;
    const bool kval1 = mk[kc + 16 + (lane & 15)] != 0;

    #pragma unroll
    for (int r = 0; r < 8; ++r) {      // online softmax per row
      float v0 = kval0 ? s[0][r] * SCALE : NEGMASK;
      float v1 = kval1 ? s[1][r] * SCALE : NEGMASK;
      float nm = fmaxf(mrun[r], half_max(fmaxf(v0, v1)));
      float corr = __expf(mrun[r] - nm);
      mrun[r] = nm;
      float p0 = __expf(v0 - nm);
      float p1 = __expf(v1 - nm);
      lrun[r] = lrun[r] * corr + half_sum(p0 + p1);
      #pragma unroll
      for (int dt = 0; dt < 4; ++dt) o[dt][r] *= corr;
      int row = r + ((lane >> 4) << 3);
      sP[wave][row * 32 +      (lane & 15)] = (__bf16)p0;
      sP[wave][row * 32 + 16 + (lane & 15)] = (__bf16)p1;
    }
    bf16x16 pa = ld_frag_a(sP[wave], 32);
    #pragma unroll
    for (int dt = 0; dt < 4; ++dt)     // P(16x32) . V(32x64)
      o[dt] = wmma_bf16(pa, ld_frag_b(sV + dt * 16 * 32, 32), o[dt]);
  }

  const int colL = lane & 15, hi = lane >> 4;
  #pragma unroll
  for (int r = 0; r < 8; ++r) {
    int row = q0 + r + hi * 8;
    float qm = mk[row] ? 1.f : 0.f;    // ctx_context *= ctx_mask (query)
    float inv = qm / lrun[r];
    #pragma unroll
    for (int dt = 0; dt < 4; ++dt)
      outp[((long)b * CTX_ + row) * E_ + h * D_ + dt * 16 + colL] =
          o[dt][r] * inv;
  }
}

// ---------------------------------------------------------------------------
// Flash cross-attention: queries = tgt (512/batch), keys = ctx ++ tgt (2560).
// tgt keys are only valid on the diagonal (no_self mask) and under tgt_mask.
// qp: [B*TGT, E], kp/vp: [B*S_TOT, E], out: [B*TGT, E]
// ---------------------------------------------------------------------------
__global__ __launch_bounds__(256) void flash_cross_kernel(
    const float* __restrict__ qp,
    const float* __restrict__ kp,
    const float* __restrict__ vp,
    const unsigned char* __restrict__ cmask,
    const unsigned char* __restrict__ tmask,
    float* __restrict__ outp)
{
  __shared__ bf16_t sK[32 * 64];
  __shared__ bf16_t sV[64 * 32];
  __shared__ bf16_t sP[8][16 * 32];

  const int tid  = threadIdx.x;
  const int lane = tid & 31;
  const int wave = tid >> 5;
  const int h = blockIdx.y, b = blockIdx.z;
  const int q0 = blockIdx.x * 128 + wave * 16;
  const float* qbase = qp + (long)b * TGT_ * E_ + h * D_;
  const float* kbase = kp + (long)b * S_TOT * E_ + h * D_;
  const float* vbase = vp + (long)b * S_TOT * E_ + h * D_;
  const unsigned char* cm = cmask + b * CTX_;
  const unsigned char* tm = tmask + b * TGT_;

  bf16x16 qa[2];
  {
    const int m = lane & 15, kh = lane >> 4;
    const float* qr = qbase + (long)(q0 + m) * E_;
    #pragma unroll
    for (int c = 0; c < 2; ++c) {
      FragU u;
      u.h[0] = cvt8(qr + c * 32 + kh * 8);
      u.h[1] = cvt8(qr + c * 32 + kh * 8 + 16);
      qa[c] = u.v;
    }
  }

  floatx8 o[4];
  #pragma unroll
  for (int dt = 0; dt < 4; ++dt)
    o[dt] = (floatx8){0.f,0.f,0.f,0.f,0.f,0.f,0.f,0.f};
  float mrun[8], lrun[8];
  #pragma unroll
  for (int r = 0; r < 8; ++r) { mrun[r] = NEGINF; lrun[r] = 0.f; }

  for (int kc = 0; kc < S_TOT; kc += 32) {
    __syncthreads();
    #pragma unroll
    for (int i = 0; i < 2; ++i) {
      int vec = tid + i * 256;
      int r   = vec >> 4;
      int c   = (vec & 15) << 2;
      float4 kv = *(const float4*)(kbase + (long)(kc + r) * E_ + c);
      st4(&sK[r * 64 + c], kv);
      float4 vv = *(const float4*)(vbase + (long)(kc + r) * E_ + c);
      sV[(c + 0) * 32 + r] = (__bf16)vv.x;
      sV[(c + 1) * 32 + r] = (__bf16)vv.y;
      sV[(c + 2) * 32 + r] = (__bf16)vv.z;
      sV[(c + 3) * 32 + r] = (__bf16)vv.w;
    }
    __syncthreads();

    floatx8 s[2];
    #pragma unroll
    for (int kt = 0; kt < 2; ++kt) {
      floatx8 a = (floatx8){0.f,0.f,0.f,0.f,0.f,0.f,0.f,0.f};
      a = wmma_bf16(qa[0], ld_frag_b(sK + kt * 16 * 64,      64), a);
      a = wmma_bf16(qa[1], ld_frag_b(sK + kt * 16 * 64 + 32, 64), a);
      s[kt] = a;
    }
    // Per-column (lane) key info for both 16-wide tiles.
    int  key[2];  bool isCtx[2], baseOk[2];
    #pragma unroll
    for (int kt = 0; kt < 2; ++kt) {
      key[kt]   = kc + kt * 16 + (lane & 15);
      isCtx[kt] = key[kt] < CTX_;
      baseOk[kt]= isCtx[kt] ? (cm[key[kt]] != 0)
                            : (tm[key[kt] - CTX_] != 0);
    }

    #pragma unroll
    for (int r = 0; r < 8; ++r) {
      int row = q0 + r + ((lane >> 4) << 3);     // global query index
      bool ok0 = baseOk[0] && (isCtx[0] || (key[0] - CTX_) == row);
      bool ok1 = baseOk[1] && (isCtx[1] || (key[1] - CTX_) == row);
      float v0 = ok0 ? s[0][r] * SCALE : NEGMASK;
      float v1 = ok1 ? s[1][r] * SCALE : NEGMASK;
      float nm = fmaxf(mrun[r], half_max(fmaxf(v0, v1)));
      float corr = __expf(mrun[r] - nm);
      mrun[r] = nm;
      float p0 = __expf(v0 - nm);
      float p1 = __expf(v1 - nm);
      lrun[r] = lrun[r] * corr + half_sum(p0 + p1);
      #pragma unroll
      for (int dt = 0; dt < 4; ++dt) o[dt][r] *= corr;
      int rloc = r + ((lane >> 4) << 3);
      sP[wave][rloc * 32 +      (lane & 15)] = (__bf16)p0;
      sP[wave][rloc * 32 + 16 + (lane & 15)] = (__bf16)p1;
    }
    bf16x16 pa = ld_frag_a(sP[wave], 32);
    #pragma unroll
    for (int dt = 0; dt < 4; ++dt)
      o[dt] = wmma_bf16(pa, ld_frag_b(sV + dt * 16 * 32, 32), o[dt]);
  }

  const int colL = lane & 15, hi = lane >> 4;
  #pragma unroll
  for (int r = 0; r < 8; ++r) {
    int row = q0 + r + hi * 8;
    float inv = 1.f / lrun[r];
    #pragma unroll
    for (int dt = 0; dt < 4; ++dt)
      outp[((long)b * TGT_ + row) * E_ + h * D_ + dt * 16 + colL] =
          o[dt][r] * inv;
  }
}

// ---------------------------------------------------------------------------
// Host-side orchestration (all launches on `stream`, graph-capture safe)
// ---------------------------------------------------------------------------
extern "C" void kernel_launch(void* const* d_in, const int* in_sizes, int n_in,
                              void* d_out, int out_size, void* d_ws, size_t ws_size,
                              hipStream_t stream) {
  (void)in_sizes; (void)n_in; (void)out_size; (void)ws_size;

  const float* ctx    = (const float*)d_in[0];
  const float* tgt    = (const float*)d_in[1];
  const unsigned char* cmask = (const unsigned char*)d_in[2];
  const unsigned char* tmask = (const unsigned char*)d_in[3];
  const float* Wqkv_w = (const float*)d_in[4];
  const float* Wqkv_b = (const float*)d_in[5];
  const float* out_w  = (const float*)d_in[6];
  const float* out_b  = (const float*)d_in[7];
  const float* cin_w  = (const float*)d_in[8];
  const float* cin_b  = (const float*)d_in[9];
  const float* cout_w = (const float*)d_in[10];
  const float* cout_b = (const float*)d_in[11];
  float* outp = (float*)d_out;

  float* ws = (float*)d_ws;
  size_t off = 0;
  float* ctx_qkv  = ws + off; off += (size_t)B_ * CTX_ * 3 * E_;   // [8192,3072]
  float* tgt_qkv  = ws + off; off += (size_t)B_ * TGT_ * 3 * E_;   // [2048,3072]
  float* ctx_attn = ws + off; off += (size_t)B_ * CTX_ * E_;       // [8192,1024]
  float* qproj    = ws + off; off += (size_t)B_ * TGT_ * E_;       // [2048,1024]
  float* kproj    = ws + off; off += (size_t)B_ * S_TOT * E_;      // [10240,1024]
  float* vproj    = ws + off; off += (size_t)B_ * S_TOT * E_;      // [10240,1024]
  float* cctx     = ws + off; off += (size_t)B_ * TGT_ * E_;       // [2048,1024]

  const dim3 blk(256);

  // 1) ctx QKV:  [8192,1024] x [3072,1024]^T
  gemm_bias_kernel<<<dim3(3 * E_ / 256, B_ * CTX_ / 128), blk, 0, stream>>>(
      ctx, (long)E_, B_ * CTX_, nullptr, 0L, 0,
      Wqkv_w, Wqkv_b, ctx_qkv, (long)(3 * E_), E_);

  // 2) tgt QKV:  [2048,1024] x [3072,1024]^T
  gemm_bias_kernel<<<dim3(3 * E_ / 256, B_ * TGT_ / 128), blk, 0, stream>>>(
      tgt, (long)E_, B_ * TGT_, nullptr, 0L, 0,
      Wqkv_w, Wqkv_b, tgt_qkv, (long)(3 * E_), E_);

  // 3) self-attention on ctx -> ctx_attn
  flash_self_kernel<<<dim3(CTX_ / 128, H_, B_), blk, 0, stream>>>(
      ctx_qkv, cmask, ctx_attn);

  // 4) ctx out-projection -> d_out[0 .. B*CTX*E)
  gemm_bias_kernel<<<dim3(E_ / 256, B_ * CTX_ / 128), blk, 0, stream>>>(
      ctx_attn, (long)E_, B_ * CTX_, nullptr, 0L, 0,
      out_w, out_b, outp, (long)E_, E_);

  // 5) cross q projection: tgt q-slice (col 0, row stride 3E) x Wq
  gemm_bias_kernel<<<dim3(E_ / 256, B_ * TGT_ / 128), blk, 0, stream>>>(
      tgt_qkv, (long)(3 * E_), B_ * TGT_, nullptr, 0L, 0,
      cin_w, cin_b, qproj, (long)E_, E_);

  // 6) cross k projection: rows = per-batch concat(ctx k-slice, tgt k-slice)
  gemm_bias_kernel<<<dim3(E_ / 256, B_ * S_TOT / 128), blk, 0, stream>>>(
      ctx_qkv + E_, (long)(3 * E_), CTX_,
      tgt_qkv + E_, (long)(3 * E_), TGT_,
      cin_w + (size_t)E_ * E_, cin_b + E_, kproj, (long)E_, E_);

  // 7) cross v projection
  gemm_bias_kernel<<<dim3(E_ / 256, B_ * S_TOT / 128), blk, 0, stream>>>(
      ctx_qkv + 2 * E_, (long)(3 * E_), CTX_,
      tgt_qkv + 2 * E_, (long)(3 * E_), TGT_,
      cin_w + (size_t)2 * E_ * E_, cin_b + 2 * E_, vproj, (long)E_, E_);

  // 8) cross attention -> cctx
  flash_cross_kernel<<<dim3(TGT_ / 128, H_, B_), blk, 0, stream>>>(
      qproj, kproj, vproj, cmask, tmask, cctx);

  // 9) tgt out-projection -> d_out[B*CTX*E ..)
  gemm_bias_kernel<<<dim3(E_ / 256, B_ * TGT_ / 128), blk, 0, stream>>>(
      cctx, (long)E_, B_ * TGT_, nullptr, 0L, 0,
      cout_w, cout_b, outp + (size_t)B_ * CTX_ * E_, (long)E_, E_);
}
